// GraphAttentionEmbedding_9715216023706
// MI455X (gfx1250) — matrix operand order repaired and checked by reading, not compiled, not run on hardware
//
#include <hip/hip_runtime.h>

// ---------------- types ----------------
typedef __attribute__((ext_vector_type(16))) _Float16 v16h;
typedef __attribute__((ext_vector_type(8)))  _Float16 v8h;
typedef __attribute__((ext_vector_type(4)))  _Float16 v4h;
typedef __attribute__((ext_vector_type(8)))  float    v8f;

union V16H { v16h v; v8h h[2]; };

#define TENC 100

// ---------------- helpers ----------------
__device__ __forceinline__ v8f wmma_f16(v16h a, v16h b, v8f c) {
    return __builtin_amdgcn_wmma_f32_16x16x32_f16(false, a, false, b,
                                                  (short)0, c, false, false);
}

// A-matrix 16x32 f16 fragment from a row-major f16 tile (stride 128).
// lanes 0-15: m=lane, K=0..15 contiguous
// lanes 16-31: m=lane-16, K=16..31 contiguous
__device__ __forceinline__ v16h load_a_frag(const _Float16* row, int kbase, int sel) {
    V16H u;
    u.h[0] = *(const v8h*)(row + kbase + sel * 16);
    u.h[1] = *(const v8h*)(row + kbase + sel * 16 + 8);
    return u.v;
}

// B-matrix 32x16 f16 fragment from W^T (row-major [n][k], stride 128):
// lane n<16 = col n holds K=kbase..+15; lanes 16-31 hold K=kbase+16..+31.
__device__ __forceinline__ v16h load_b_frag(const _Float16* wT, int nbase, int kbase, int lane) {
    int col = lane & 15;
    int sel = lane >> 4;
    return *(const v16h*)(wT + (size_t)(nbase + col) * 128 + kbase + sel * 16);
}

// monotone float<->uint map for atomicMax on floats
__device__ __forceinline__ unsigned fmapu(float x) {
    unsigned b = __float_as_uint(x);
    return (b & 0x80000000u) ? ~b : (b | 0x80000000u);
}
__device__ __forceinline__ float funmap(unsigned u) {
    unsigned b = (u & 0x80000000u) ? (u ^ 0x80000000u) : ~u;
    return __uint_as_float(b);
}

// ---------------- kernel 0: weight prep (f16, transposed) ----------------
// wT5[m*16384 + n*128 + k] = W[k][n]; m in {0:Wq,1:Wk,2:Wv,3:Ws,4:We(zero-pad k>=100)}
__global__ void prep_weights(const float* __restrict__ Wq, const float* __restrict__ Wk,
                             const float* __restrict__ Wv, const float* __restrict__ Ws,
                             const float* __restrict__ We, _Float16* __restrict__ wT5) {
    int i = blockIdx.x * blockDim.x + threadIdx.x;
    if (i >= 5 * 16384) return;
    int m = i >> 14;
    int rem = i & 16383;
    int n = rem >> 7;
    int k = rem & 127;
    float val;
    if (m == 4) {
        val = (k < TENC) ? We[k * 128 + n] : 0.0f;
    } else {
        const float* W = (m == 0) ? Wq : (m == 1) ? Wk : (m == 2) ? Wv : Ws;
        val = W[k * 128 + n];
    }
    wT5[i] = (_Float16)val;
}

// ---------------- kernel 1: fused node GEMMs via WMMA ----------------
// One block = one 64-row x-tile (loaded ONCE via async-to-LDS), all 4 weight
// matrices computed per block (x traffic 4x lower than one-matrix-per-block).
__global__ __launch_bounds__(256) void node_gemm(
    const float* __restrict__ x, const _Float16* __restrict__ wT5,
    const float* __restrict__ bq, const float* __restrict__ bk,
    const float* __restrict__ bv, const float* __restrict__ bs,
    float* __restrict__ Q, float* __restrict__ K, float* __restrict__ V,
    float* __restrict__ Sout, int Nn) {

    __shared__ __align__(16) float    xstage[64 * 128];   // 32 KB f32 staging
    __shared__ __align__(32) _Float16 xs[64 * 128];       // 16 KB f16 WMMA tile
    const int tid = threadIdx.x;
    const int rbase = blockIdx.x * 64;

    // ---- async copy of the 64x128 f32 tile into LDS (ASYNCcnt path) ----
    // LDS offset = low 32 bits of the flat shared address (aperture math).
    unsigned ldsbase = (unsigned)(uintptr_t)&xstage[0];
    unsigned long long xbase = (unsigned long long)(uintptr_t)x;
    #pragma unroll
    for (int kk = 0; kk < 8; ++kk) {
        int c = tid + kk * 256;           // 16-byte chunk id, 0..2047
        int row = c >> 5;                 // 32 chunks per 512B row
        int col16 = c & 31;
        int grow = rbase + row;
        if (grow >= Nn) grow = Nn - 1;    // clamp tail rows
        unsigned long long gaddr = xbase + (unsigned long long)grow * 512ull
                                         + (unsigned long long)(col16 * 16);
        unsigned loff = ldsbase + (unsigned)c * 16u;
        asm volatile("global_load_async_to_lds_b128 %0, %1, off"
                     :: "v"(loff), "v"(gaddr) : "memory");
    }
    asm volatile("s_wait_asynccnt 0x0" ::: "memory");
    __syncthreads();

    // ---- f32 -> f16 conversion in LDS ----
    for (int i = tid; i < 64 * 128; i += 256) xs[i] = (_Float16)xstage[i];
    __syncthreads();

    const int wave = tid >> 5, lane = tid & 31;
    const int mtile = wave >> 1;          // rows mtile*16..+15
    const int nhalf = (wave & 1) * 64;    // 0 or 64
    const int sel = lane >> 4, nl = lane & 15;
    const _Float16* arow = &xs[(size_t)(mtile * 16 + nl) * 128];

    #pragma unroll
    for (int w = 0; w < 4; ++w) {
        const _Float16* wT = wT5 + (size_t)w * 16384;
        v8f acc[4];
        #pragma unroll
        for (int nt = 0; nt < 4; ++nt) acc[nt] = (v8f){};

        #pragma unroll
        for (int kb = 0; kb < 4; ++kb) {
            v16h a = load_a_frag(arow, kb * 32, sel);
            #pragma unroll
            for (int nt = 0; nt < 4; ++nt) {
                v16h b = load_b_frag(wT, nhalf + nt * 16, kb * 32, lane);
                acc[nt] = wmma_f16(a, b, acc[nt]);
            }
        }

        const float* bias = (w == 0) ? bq : (w == 1) ? bk : (w == 2) ? bv : bs;
        float* dst = (w == 0) ? Q : (w == 1) ? K : (w == 2) ? V : Sout;
        #pragma unroll
        for (int nt = 0; nt < 4; ++nt) {
            int n = nhalf + nt * 16 + nl;
            float bval = bias[n];
            #pragma unroll
            for (int r = 0; r < 8; ++r) {
                int row = rbase + mtile * 16 + sel * 8 + r;
                if (row < Nn) dst[(size_t)row * 128 + n] = acc[nt][r] + bval;
            }
        }
    }
}

// ---------------- kernel 2: zero max/sum buffers ----------------
__global__ void init_ms(unsigned* __restrict__ mbuf, float* __restrict__ sbuf, int n2) {
    int i = blockIdx.x * blockDim.x + threadIdx.x;
    if (i < n2) { mbuf[i] = 0u; sbuf[i] = 0.0f; }
}

// ---------------- kernel 3: time encoding + edge GEMM via WMMA ----------------
__global__ __launch_bounds__(256) void edge_time_gemm(
    const int* __restrict__ ei, const float* __restrict__ last_update,
    const float* __restrict__ t, const float* __restrict__ Wt,
    const float* __restrict__ bt, const _Float16* __restrict__ weT,
    _Float16* __restrict__ eproj, int Ee) {

    __shared__ float rel[128];
    __shared__ __align__(32) _Float16 attr[128 * 128];   // 32 KB
    const int tid = threadIdx.x;
    const int ebase = blockIdx.x * 128;

    if (tid < 128) {
        int e = ebase + tid;
        if (e >= Ee) e = Ee - 1;
        int s_ = ei[e];
        rel[tid] = last_update[s_] - t[e];
    }
    __syncthreads();

    for (int i = tid; i < 128 * 128; i += 256) {
        int el = i >> 7, j = i & 127;
        float v = 0.0f;
        if (j < TENC) v = cosf(rel[el] * Wt[j] + bt[j]);
        attr[i] = (_Float16)v;
    }
    __syncthreads();

    const int wave = tid >> 5, lane = tid & 31;
    const int sel = lane >> 4, nl = lane & 15;
    const _Float16* arow = &attr[(size_t)(wave * 16 + nl) * 128];

    v8f acc[8];
    #pragma unroll
    for (int nt = 0; nt < 8; ++nt) acc[nt] = (v8f){};

    #pragma unroll
    for (int kb = 0; kb < 4; ++kb) {
        v16h a = load_a_frag(arow, kb * 32, sel);
        #pragma unroll
        for (int nt = 0; nt < 8; ++nt) {
            v16h b = load_b_frag(weT, nt * 16, kb * 32, lane);
            acc[nt] = wmma_f16(a, b, acc[nt]);
        }
    }

    #pragma unroll
    for (int nt = 0; nt < 8; ++nt) {
        int n = nt * 16 + nl;
        #pragma unroll
        for (int r = 0; r < 8; ++r) {
            int e = ebase + wave * 16 + sel * 8 + r;
            if (e < Ee) eproj[(size_t)e * 128 + n] = (_Float16)acc[nt][r];
        }
    }
}

// ---------------- kernel 4: attention logits + segment max ----------------
__global__ __launch_bounds__(256) void alpha_kernel(
    const int* __restrict__ ei, const float* __restrict__ Q,
    const float* __restrict__ K, const _Float16* __restrict__ eproj,
    float* __restrict__ alpha, unsigned* __restrict__ mbuf, int Ee) {

    int gtid = blockIdx.x * blockDim.x + threadIdx.x;
    int e = gtid >> 5;
    int lane = threadIdx.x & 31;
    if (e >= Ee) return;

    int src = ei[e];
    int dst = ei[Ee + e];
    int c0 = lane * 4;

    float4 qv = *(const float4*)(Q + (size_t)dst * 128 + c0);
    float4 kv = *(const float4*)(K + (size_t)src * 128 + c0);
    v4h  eh  = *(const v4h*)(eproj + (size_t)e * 128 + c0);

    float p = qv.x * (kv.x + (float)eh[0]) + qv.y * (kv.y + (float)eh[1]) +
              qv.z * (kv.z + (float)eh[2]) + qv.w * (kv.w + (float)eh[3]);
    for (int off = 8; off; off >>= 1) p += __shfl_xor(p, off, 16);
    p *= 0.125f;   // 1/sqrt(64)

    if ((lane & 15) == 0) {
        int h = lane >> 4;
        alpha[(size_t)e * 2 + h] = p;
        atomicMax(mbuf + (size_t)dst * 2 + h, fmapu(p));
    }
}

// ---------------- kernel 5: segment sum of exp ----------------
__global__ void sum_kernel(const int* __restrict__ ei, const float* __restrict__ alpha,
                           const unsigned* __restrict__ mbuf, float* __restrict__ sbuf,
                           int Ee) {
    int i = blockIdx.x * blockDim.x + threadIdx.x;
    if (i >= Ee * 2) return;
    int e = i >> 1, h = i & 1;
    int dst = ei[Ee + e];
    float a = expf(alpha[i] - funmap(mbuf[(size_t)dst * 2 + h]));
    atomicAdd(sbuf + (size_t)dst * 2 + h, a);
}

// ---------------- kernel 6: weighted scatter-add ----------------
__global__ __launch_bounds__(256) void aggregate(
    const int* __restrict__ ei, const float* __restrict__ V,
    const _Float16* __restrict__ eproj, const float* __restrict__ alpha,
    const unsigned* __restrict__ mbuf, const float* __restrict__ sbuf,
    float* __restrict__ out, int Ee) {

    int gtid = blockIdx.x * blockDim.x + threadIdx.x;
    int e = gtid >> 5;
    int lane = threadIdx.x & 31;
    if (e >= Ee) return;

    int src = ei[e];
    int dst = ei[Ee + e];
    int h = lane >> 4;

    float mh = funmap(mbuf[(size_t)dst * 2 + h]);
    float sh = sbuf[(size_t)dst * 2 + h];
    float a  = expf(alpha[(size_t)e * 2 + h] - mh) / (sh + 1e-16f);

    int c0 = lane * 4;
    float4 vv = *(const float4*)(V + (size_t)src * 128 + c0);
    v4h  eh  = *(const v4h*)(eproj + (size_t)e * 128 + c0);

    float* o = out + (size_t)dst * 128 + c0;
    atomicAdd(o + 0, a * (vv.x + (float)eh[0]));
    atomicAdd(o + 1, a * (vv.y + (float)eh[1]));
    atomicAdd(o + 2, a * (vv.z + (float)eh[2]));
    atomicAdd(o + 3, a * (vv.w + (float)eh[3]));
}

// ---------------- host launcher ----------------
extern "C" void kernel_launch(void* const* d_in, const int* in_sizes, int n_in,
                              void* d_out, int out_size, void* d_ws, size_t ws_size,
                              hipStream_t stream) {
    const float* x   = (const float*)d_in[0];
    const float* lu  = (const float*)d_in[1];
    const float* t   = (const float*)d_in[2];
    // d_in[3] = msg, empty
    const int*   ei  = (const int*)d_in[4];
    const float* Wt  = (const float*)d_in[5];
    const float* bt  = (const float*)d_in[6];
    const float* Wq  = (const float*)d_in[7];
    const float* bq  = (const float*)d_in[8];
    const float* Wk  = (const float*)d_in[9];
    const float* bk  = (const float*)d_in[10];
    const float* Wv  = (const float*)d_in[11];
    const float* bv  = (const float*)d_in[12];
    const float* We  = (const float*)d_in[13];
    const float* Ws  = (const float*)d_in[14];
    const float* bs  = (const float*)d_in[15];
    float* out = (float*)d_out;

    const int N = in_sizes[0] / 128;
    const int E = in_sizes[2];

    auto align256 = [](size_t v) { return (v + 255) & ~(size_t)255; };
    char* w = (char*)d_ws;
    size_t off = 0;
    _Float16* wT5   = (_Float16*)(w + off); off = align256(off + (size_t)5 * 16384 * sizeof(_Float16));
    float*    Qb    = (float*)(w + off);    off = align256(off + (size_t)N * 128 * sizeof(float));
    float*    Kb    = (float*)(w + off);    off = align256(off + (size_t)N * 128 * sizeof(float));
    float*    Vb    = (float*)(w + off);    off = align256(off + (size_t)N * 128 * sizeof(float));
    _Float16* eproj = (_Float16*)(w + off); off = align256(off + (size_t)E * 128 * sizeof(_Float16));
    float*    alpha = (float*)(w + off);    off = align256(off + (size_t)E * 2 * sizeof(float));
    unsigned* mbuf  = (unsigned*)(w + off); off = align256(off + (size_t)N * 2 * sizeof(unsigned));
    float*    sbuf  = (float*)(w + off);    off = align256(off + (size_t)N * 2 * sizeof(float));
    (void)ws_size; (void)n_in; (void)out_size;

    // 0) weights -> f16 transposed
    prep_weights<<<(5 * 16384 + 255) / 256, 256, 0, stream>>>(Wq, Wk, Wv, Ws, We, wT5);

    // 1) fused node GEMMs (Q,K,V to ws; skip path directly into d_out)
    node_gemm<<<(N + 63) / 64, 256, 0, stream>>>(x, wT5, bq, bk, bv, bs, Qb, Kb, Vb, out, N);

    // 2) zero per-(node,head) max/sum
    init_ms<<<(N * 2 + 255) / 256, 256, 0, stream>>>(mbuf, sbuf, N * 2);

    // 3) time encoding + edge projection GEMM
    edge_time_gemm<<<(E + 127) / 128, 256, 0, stream>>>(ei, lu, t, Wt, bt,
                                                        wT5 + (size_t)4 * 16384, eproj, E);

    // 4) attention logits + segment max
    long long threads4 = (long long)E * 32;
    alpha_kernel<<<(unsigned)((threads4 + 255) / 256), 256, 0, stream>>>(
        ei, Qb, Kb, eproj, alpha, mbuf, E);

    // 5) segment sum of exp
    sum_kernel<<<(E * 2 + 255) / 256, 256, 0, stream>>>(ei, alpha, mbuf, sbuf, E);

    // 6) weighted scatter-add into d_out (already holds skip connection)
    aggregate<<<(unsigned)((threads4 + 255) / 256), 256, 0, stream>>>(
        ei, Vb, eproj, alpha, mbuf, sbuf, out, E);
}
